// QuantumWalkRetriever_86543591014915
// MI455X (gfx1250) — matrix operand order repaired and checked by reading, not compiled, not run on hardware
//
#include <hip/hip_runtime.h>
#include <math.h>

// Problem constants (match reference)
#define Bsz   16
#define Nn    8192
#define Dd    384
#define DEGc  16
#define Kk    8
#define Hh    128
#define TWO_D 768
#define CTOT  256          // [W1 cols | P1 cols]
#define HB_STR 136         // LDS row stride (bf16 elements), 128 + pad

typedef __bf16 bf16_t;
typedef __attribute__((ext_vector_type(16))) __bf16 v16bf;
typedef __attribute__((ext_vector_type(8)))  __bf16 v8bf;
typedef __attribute__((ext_vector_type(8)))  float  v8f;
typedef __attribute__((ext_vector_type(4)))  float  v4f;

// ---------------- prep: pack weights transposed to bf16 ----------------
// Wt[c][f] = (c<128 ? W1[f][c] : P1[f][c-128]),  c in [0,256), f in [0,768)
__global__ void pack_weights(const float* __restrict__ W1, const float* __restrict__ P1,
                             bf16_t* __restrict__ Wt) {
  int idx = blockIdx.x * 256 + threadIdx.x;
  if (idx >= CTOT * TWO_D) return;
  int c = idx / TWO_D, f = idx % TWO_D;
  float v = (c < Hh) ? W1[f * Hh + c] : P1[f * Hh + (c - Hh)];
  Wt[idx] = (bf16_t)v;
}

// B2t[c][k]: stage-2 weight, 16 cols x 256 rows (transposed storage).
// cols 0..7 = W2 (rows k<128), col 8 = P2 (rows k>=128), else 0.
__global__ void pack_b2(const float* __restrict__ W2, const float* __restrict__ P2,
                        bf16_t* __restrict__ B2t) {
  int idx = blockIdx.x * 256 + threadIdx.x;
  if (idx >= 16 * CTOT) return;
  int c = idx / CTOT, k = idx % CTOT;
  float v = 0.f;
  if (c < 8)        { if (k < Hh)  v = W2[k * Kk + c]; }
  else if (c == 8)  { if (k >= Hh) v = P2[k - Hh];     }
  B2t[idx] = (bf16_t)v;
}

__global__ void pack_bias(const float* __restrict__ b1, const float* __restrict__ pb1,
                          float* __restrict__ bcat) {
  int c = threadIdx.x;                      // 256 threads
  bcat[c] = (c < Hh) ? b1[c] : pb1[c - Hh];
}

// ---------------- fused MLP -> amps, WMMA bf16 ----------------
// One wave per 16-node tile. Stage1: h = relu(inp @ [W1|P1] + bias) via
// 2 passes x 8 col-tiles x 24 K-steps of v_wmma_f32_16x16x32_bf16.
// Stage2: D2 = h(16x256) @ B2(256x16) -> cols 0..7 = amp, col 8 = path logit.
__launch_bounds__(256)
__global__ void mlp_amps(const float* __restrict__ emb, const float* __restrict__ q,
                         const bf16_t* __restrict__ Wt, const float* __restrict__ bcat,
                         const bf16_t* __restrict__ B2t, const float* __restrict__ b2,
                         const float* __restrict__ pb2v, const int* __restrict__ labels,
                         float* __restrict__ amps) {
  __shared__ bf16_t hb[8][16][HB_STR];      // per-wave 16x128 bf16 h slice

  const int lane = threadIdx.x & 31;
  const int wave = threadIdx.x >> 5;
  const int tile = blockIdx.x * 8 + wave;   // 8192 tiles total
  const int node0 = tile * 16;
  const int b  = node0 >> 13;               // N = 8192
  const int n0 = node0 & (Nn - 1);
  const int m    = lane & 15;
  const int half = lane >> 4;

  v8f acc2; 
  #pragma unroll
  for (int i = 0; i < 8; ++i) acc2[i] = 0.f;

  for (int pass = 0; pass < 2; ++pass) {
    v8f acc[8];
    #pragma unroll
    for (int t = 0; t < 8; ++t)
      #pragma unroll
      for (int i = 0; i < 8; ++i) acc[t][i] = 0.f;

    for (int kk = 0; kk < 24; ++kk) {
      const int f0 = kk * 32;
      // ---- A tile: 16 nodes x 32 features, fp32 -> bf16 ----
      const float* rp;
      if (f0 < Dd) rp = emb + ((size_t)b * Nn + n0 + m) * Dd + f0;
      else         rp = q + (size_t)b * Dd + (f0 - Dd);      // q rows identical
      v4f a0 = *(const v4f*)(rp + half * 8);
      v4f a1 = *(const v4f*)(rp + half * 8 + 4);
      v4f a2 = *(const v4f*)(rp + 16 + half * 8);
      v4f a3 = *(const v4f*)(rp + 16 + half * 8 + 4);
      v16bf A;
      #pragma unroll
      for (int j = 0; j < 4; ++j) {
        A[j]      = (bf16_t)a0[j];
        A[4 + j]  = (bf16_t)a1[j];
        A[8 + j]  = (bf16_t)a2[j];
        A[12 + j] = (bf16_t)a3[j];
      }
      // ---- 8 column tiles ----
      #pragma unroll
      for (int t = 0; t < 8; ++t) {
        const int c = pass * 128 + t * 16 + m;               // weight column
        const bf16_t* wp = Wt + (size_t)c * TWO_D + f0 + half * 8;
        v8bf blo = *(const v8bf*)(wp);
        v8bf bhi = *(const v8bf*)(wp + 16);
        v16bf Bv;
        #pragma unroll
        for (int j = 0; j < 8; ++j) { Bv[j] = blo[j]; Bv[8 + j] = bhi[j]; }
        acc[t] = __builtin_amdgcn_wmma_f32_16x16x32_bf16(
            false, A, false, Bv, (short)0, acc[t], false, false);
      }
    }

    // bias + relu -> LDS (C-layout: lane = col, reg r = row r + 8*half)
    #pragma unroll
    for (int t = 0; t < 8; ++t) {
      const int cl = t * 16 + m;                // local col 0..127
      const float bias = bcat[pass * 128 + cl];
      #pragma unroll
      for (int r = 0; r < 8; ++r) {
        float v = acc[t][r] + bias;
        v = v > 0.f ? v : 0.f;
        hb[wave][r + half * 8][cl] = (bf16_t)v;
      }
    }

    // stage-2 partial: h_slice(16x128) @ B2 rows [pass*128 .. +128)
    for (int kk = 0; kk < 4; ++kk) {
      const int kl = kk * 32;                   // local K
      const bf16_t* hp = &hb[wave][m][kl];
      v8bf lo = *(const v8bf*)(hp + half * 8);
      v8bf hi = *(const v8bf*)(hp + 16 + half * 8);
      v16bf A2;
      #pragma unroll
      for (int j = 0; j < 8; ++j) { A2[j] = lo[j]; A2[8 + j] = hi[j]; }
      const int kg = pass * 128 + kl;
      const bf16_t* bp = B2t + (size_t)m * CTOT + kg + half * 8;
      v8bf blo = *(const v8bf*)(bp);
      v8bf bhi = *(const v8bf*)(bp + 16);
      v16bf Bv;
      #pragma unroll
      for (int j = 0; j < 8; ++j) { Bv[j] = blo[j]; Bv[8 + j] = bhi[j]; }
      acc2 = __builtin_amdgcn_wmma_f32_16x16x32_bf16(
          false, A2, false, Bv, (short)0, acc2, false, false);
    }
  }

  // path = sigmoid(col8 + pb2) + label; amps = (col_k + b2[k]) * (1 + path)
  const float pb2 = pb2v[0];
  const int c = lane & 15;
  #pragma unroll
  for (int r = 0; r < 8; ++r) {
    const int node = n0 + r + half * 8;
    const float plog = __shfl(acc2[r], (half << 4) | 8, 32);  // broadcast col 8
    const float path = 1.f / (1.f + __expf(-(plog + pb2)));
    const float scale = 1.f + path + (float)labels[(size_t)b * Nn + node];
    if (c < 8) {
      const float av = (acc2[r] + b2[c]) * scale;
      amps[((size_t)b * Nn + node) * Kk + c] = av;
    }
  }
}

// ---------------- diffusion pass (unnormalized; norm deferred) ----------------
// dst[b,i,k] = scale * sum_d src[b, nb[b,i,d], k]  (* amps[b,i,k] if mulAmps)
__global__ void spmm_step(const float* __restrict__ src, const float* __restrict__ amps,
                          const int* __restrict__ nbrs, float* __restrict__ dst,
                          int mulAmps, float scale) {
  const int gid = blockIdx.x * blockDim.x + threadIdx.x;   // B*N*K
  if (gid >= Bsz * Nn * Kk) return;
  const int k = gid & 7;
  const int i = (gid >> 3) & (Nn - 1);
  const int b = gid >> 16;
  const int* nb = nbrs + ((size_t)b * Nn + i) * DEGc;
  const float* sb = src + (size_t)b * Nn * Kk;
  float s = 0.f;
  #pragma unroll
  for (int d = 0; d < DEGc; ++d) {
    const int j = nb[d];
    s += sb[(size_t)j * Kk + k];
  }
  s *= scale;
  if (mulAmps) s *= amps[gid];
  dst[gid] = s;
}

// ---------------- deterministic per-batch L2 norm ----------------
__global__ void norm_kernel(const float* __restrict__ st, float* __restrict__ rnorm) {
  __shared__ float red[1024];
  const int b = blockIdx.x;
  const float* p = st + (size_t)b * Nn * Kk;
  float s = 0.f;
  for (int idx = threadIdx.x; idx < Nn * Kk; idx += 1024) {
    const float v = p[idx];
    s += v * v;
  }
  red[threadIdx.x] = s;
  __syncthreads();
  for (int off = 512; off > 0; off >>= 1) {
    if (threadIdx.x < off) red[threadIdx.x] += red[threadIdx.x + off];
    __syncthreads();
  }
  if (threadIdx.x == 0) {
    const float n2 = red[0];
    rnorm[b] = (n2 > 0.f) ? rsqrtf(n2) : 1.f;
  }
}

// ---------------- output: sum_k |st| * rnorm ----------------
__global__ void out_kernel(const float* __restrict__ st, const float* __restrict__ rnorm,
                           float* __restrict__ out) {
  const int gid = blockIdx.x * 256 + threadIdx.x;   // B*N
  if (gid >= Bsz * Nn) return;
  const int b = gid >> 13;
  const v4f* p = (const v4f*)(st + (size_t)gid * Kk);
  const v4f x = p[0], y = p[1];
  const float s = fabsf(x[0]) + fabsf(x[1]) + fabsf(x[2]) + fabsf(x[3]) +
                  fabsf(y[0]) + fabsf(y[1]) + fabsf(y[2]) + fabsf(y[3]);
  out[gid] = s * rnorm[b];
}

extern "C" void kernel_launch(void* const* d_in, const int* in_sizes, int n_in,
                              void* d_out, int out_size, void* d_ws, size_t ws_size,
                              hipStream_t stream) {
  (void)in_sizes; (void)n_in; (void)out_size; (void)ws_size;
  const float* q      = (const float*)d_in[0];
  const float* emb    = (const float*)d_in[1];
  const int*   nbrs   = (const int*)  d_in[2];
  const int*   labels = (const int*)  d_in[3];
  const float* W1     = (const float*)d_in[4];
  const float* b1     = (const float*)d_in[5];
  const float* W2     = (const float*)d_in[6];
  const float* b2     = (const float*)d_in[7];
  const float* P1     = (const float*)d_in[8];
  const float* pb1    = (const float*)d_in[9];
  const float* P2     = (const float*)d_in[10];
  const float* pb2    = (const float*)d_in[11];
  float* out = (float*)d_out;

  char* ws = (char*)d_ws;
  size_t off = 0;
  auto alloc = [&](size_t bytes) -> void* {
    void* p = ws + off;
    off += (bytes + 255) & ~(size_t)255;
    return p;
  };
  const size_t BNK = (size_t)Bsz * Nn * Kk;
  float*  amps = (float*) alloc(BNK * 4);
  float*  bufA = (float*) alloc(BNK * 4);
  float*  bufB = (float*) alloc(BNK * 4);
  bf16_t* Wt   = (bf16_t*)alloc((size_t)CTOT * TWO_D * 2);
  bf16_t* B2t  = (bf16_t*)alloc((size_t)16 * CTOT * 2);
  float*  bcat = (float*) alloc(CTOT * 4);
  float*  rnorm= (float*) alloc(Bsz * 4);

  // prep (weights are tiny; stay L2-hot for the GEMM)
  pack_weights<<<(CTOT * TWO_D + 255) / 256, 256, 0, stream>>>(W1, P1, Wt);
  pack_b2<<<(16 * CTOT + 255) / 256, 256, 0, stream>>>(W2, P2, B2t);
  pack_bias<<<1, 256, 0, stream>>>(b1, pb1, bcat);

  // fused MLP -> amps (WMMA bf16)
  mlp_amps<<<1024, 256, 0, stream>>>(emb, q, Wt, bcat, B2t, b2, pb2, labels, amps);

  // state3 = M^3 s0 / ||M^3 s0||; s0 uniform c0 folded into pass 1.
  const float c0 = 1.0f / 256.0f;   // 1/sqrt(N*K)
  spmm_step<<<4096, 256, 0, stream>>>(amps, amps, nbrs, bufA, 1, c0);   // prod1
  spmm_step<<<4096, 256, 0, stream>>>(bufA, amps, nbrs, bufB, 1, 1.f);  // prod2
  spmm_step<<<4096, 256, 0, stream>>>(bufB, amps, nbrs, bufA, 0, 1.f);  // st3

  norm_kernel<<<Bsz, 1024, 0, stream>>>(bufA, rnorm);
  out_kernel<<<(Bsz * Nn + 255) / 256, 256, 0, stream>>>(bufA, rnorm, out);
}
